// GatedLinearAttention_82592221102859
// MI455X (gfx1250) — compile-verified, matrix operand order checked by reading, not run on hardware
//
#include <hip/hip_runtime.h>

typedef __attribute__((ext_vector_type(16))) __bf16 v16bf;
typedef __attribute__((ext_vector_type(8)))  float  v8f;
typedef __attribute__((ext_vector_type(4)))  unsigned int v4u;
typedef __attribute__((ext_vector_type(8)))  int v8i;
typedef __attribute__((ext_vector_type(4)))  int v4i;

#define C_DIM 128
#define NPIX  16384   // H*W
#define BATCH 16

#if __has_builtin(__builtin_amdgcn_tensor_load_to_lds)
#define HAVE_TDM 1
#else
#define HAVE_TDM 0
#endif

// ---- workspace map (units: unsigned short elements unless noted) ----
// 0       : WkA   (16384)   bf16, A-operand swizzled
// 16384   : WvA   (16384)
// 32768   : WgA   (32768)   rows 0..127 = forget, 128..255 = input
// 65536   : WqA   (16384)
// 81920   : WpA   (16384)
// byte 196608 : know  f32 (B*C*C)          -- zeroed each launch
// byte 1245184: fsum  f32 (B*C)            -- zeroed each launch
// byte 1253376: memA  bf16 (B*C*C)  new_memory^T, A-operand swizzled
#define KNOW_BYTE 196608
#define FSUM_BYTE 1245184
#define MEMA_BYTE 1253376

__device__ __forceinline__ unsigned short f2bf(float f) {
  union { float f; unsigned int u; } v; v.f = f;
  unsigned int u = v.u;
  u += 0x7FFFu + ((u >> 16) & 1u);          // round-to-nearest-even
  return (unsigned short)(u >> 16);
}

// element index inside a 512-element 16(M/col) x 32(K) WMMA operand tile,
// stored so that each lane's 16 bf16 values are contiguous (32B/lane).
// Layout per ISA 7.12.2 (16-bit A 16x32 / mirrored for B 32x16):
//   lane = (m&15) | (((k>>3)&1)<<4) ;  elem = (k&7) | ((k>=16)<<3)
__device__ __forceinline__ int swz(int m16, int k32) {
  int lane = (m16 & 15) | (((k32 >> 3) & 1) << 4);
  int e    = (k32 & 7) | ((k32 >> 4) << 3);
  return lane * 16 + e;
}

// fast sigmoid: v_exp_f32 + v_rcp_f32 (avoid the IEEE div expansion)
__device__ __forceinline__ float sigmoidf_(float x) {
  return __builtin_amdgcn_rcpf(1.0f + __expf(-x));
}

__device__ __forceinline__ v8f wmma_bf16(v16bf a, v16bf b, v8f c) {
  return __builtin_amdgcn_wmma_f32_16x16x32_bf16(false, a, false, b, (short)0, c,
                                                 false, false);
}

// ---------------------------------------------------------------------------
// k0: convert all weight matrices to bf16 in A-operand swizzled tile order
// ---------------------------------------------------------------------------
__global__ void __launch_bounds__(256) glak0_prep(
    const float* __restrict__ Wq, const float* __restrict__ Wk,
    const float* __restrict__ Wv, const float* __restrict__ Wg,
    const float* __restrict__ Wp, unsigned short* __restrict__ ws16)
{
  int idx = blockIdx.x * 256 + threadIdx.x;     // 98304 total elements
  const float* src; int dst, local;
  if (idx < 16384)      { src = Wk; dst = 0;     local = idx;          }
  else if (idx < 32768) { src = Wv; dst = 16384; local = idx - 16384;  }
  else if (idx < 65536) { src = Wg; dst = 32768; local = idx - 32768;  }
  else if (idx < 81920) { src = Wq; dst = 65536; local = idx - 65536;  }
  else                  { src = Wp; dst = 81920; local = idx - 81920;  }
  int o = local >> 7, c = local & 127;
  ws16[dst + ((o >> 4) * 4 + (c >> 5)) * 512 + swz(o & 15, c & 31)] = f2bf(src[local]);
}

// ---------------------------------------------------------------------------
// k1: k/v/gates projections + gating + knowledge partials + forget sums.
// One block = one batch x one 256-column strip (4 chunks of 64 cols).
// 8 waves; wave w owns output row-tile w (channels w*16..w*16+15).
// ---------------------------------------------------------------------------
__global__ void __launch_bounds__(256) glak1_proj_knowledge(
    const float* __restrict__ x, const unsigned short* __restrict__ ws16,
    const float* __restrict__ bk, const float* __restrict__ bv,
    const float* __restrict__ bg, float* __restrict__ know,
    float* __restrict__ fsum)
{
  __shared__ alignas(64) unsigned short lds[24576];   // 48 KB
  unsigned short* lx  = lds;          // x  bf16, B-order: tiles (kt:4, nt:4)
  unsigned short* lkg = lds + 8192;   // kg bf16, A-order: tiles (rt:8, kt:2)
  unsigned short* lv  = lds + 16384;  // v  bf16, B-order: tiles (kt:2, dt:8)

  const int tid  = threadIdx.x;
  const int lane = tid & 31;
  const int w    = tid >> 5;
  const int hl   = lane >> 4;
  const int nl   = lane & 15;

  const int b   = blockIdx.x >> 6;    // 64 blocks per batch
  const int blk = blockIdx.x & 63;

  const unsigned short* wkA = ws16;
  const unsigned short* wvA = ws16 + 16384;
  const unsigned short* wgA = ws16 + 32768;

  v8f zero = {};
  v8f kacc[8];
  #pragma unroll
  for (int d = 0; d < 8; ++d) kacc[d] = zero;
  float fs[8];
  #pragma unroll
  for (int i = 0; i < 8; ++i) fs[i] = 0.0f;

  float bk_r[8], bv_r[8], bgF_r[8], bgI_r[8];
  #pragma unroll
  for (int i = 0; i < 8; ++i) {
    int o = w * 16 + hl * 8 + i;
    bk_r[i]  = bk[o];
    bv_r[i]  = bv[o];
    bgF_r[i] = bg[o];
    bgI_r[i] = bg[C_DIM + o];
  }

  for (int ch = 0; ch < 4; ++ch) {
    const int n0 = (blk * 4 + ch) * 64;
    // stage x chunk (128 ch x 64 cols) to LDS, bf16, B-operand order
    for (int idx = tid; idx < C_DIM * 64; idx += 256) {
      int c = idx >> 6, n = idx & 63;
      float xv = x[(size_t)(b * C_DIM + c) * NPIX + n0 + n];
      lx[((c >> 5) * 4 + (n >> 4)) * 512 + swz(n & 15, c & 31)] = f2bf(xv);
    }
    // prefetch next chunk's x strip into cache while we compute this one
    if (ch < 3) {
      const float* nx = x + (size_t)(b * C_DIM) * NPIX + (size_t)(blk * 4 + ch + 1) * 64;
      int r = tid >> 1, half = tid & 1;
      __builtin_prefetch(nx + (size_t)r * NPIX + half * 32, 0, 1);
    }
    __syncthreads();

    // ---- input gate (Wg rows C..2C): keep sigmoid in registers ----
    v8f sig[4];
    #pragma unroll
    for (int ct = 0; ct < 4; ++ct) {
      v8f acc = zero;
      #pragma unroll
      for (int kk = 0; kk < 4; ++kk) {
        v16bf a  = *(const v16bf*)(wgA + ((8 + w) * 4 + kk) * 512 + lane * 16);
        v16bf bm = *(const v16bf*)(lx + (kk * 4 + ct) * 512 + lane * 16);
        acc = wmma_bf16(a, bm, acc);
      }
      #pragma unroll
      for (int i = 0; i < 8; ++i) sig[ct][i] = sigmoidf_(acc[i] + bgI_r[i]);
    }
    // ---- k, gated -> lkg (A-order: M=channel, K=col) ----
    #pragma unroll
    for (int ct = 0; ct < 4; ++ct) {
      v8f acc = zero;
      #pragma unroll
      for (int kk = 0; kk < 4; ++kk) {
        v16bf a  = *(const v16bf*)(wkA + (w * 4 + kk) * 512 + lane * 16);
        v16bf bm = *(const v16bf*)(lx + (kk * 4 + ct) * 512 + lane * 16);
        acc = wmma_bf16(a, bm, acc);
      }
      #pragma unroll
      for (int i = 0; i < 8; ++i) {
        float kg = (acc[i] + bk_r[i]) * sig[ct][i];
        int K = ct * 16 + nl;
        lkg[(w * 2 + (K >> 5)) * 512 + swz(hl * 8 + i, K & 31)] = f2bf(kg);
      }
    }
    // ---- v -> lv (B-order: K=col, col=channel d) ----
    #pragma unroll
    for (int ct = 0; ct < 4; ++ct) {
      v8f acc = zero;
      #pragma unroll
      for (int kk = 0; kk < 4; ++kk) {
        v16bf a  = *(const v16bf*)(wvA + (w * 4 + kk) * 512 + lane * 16);
        v16bf bm = *(const v16bf*)(lx + (kk * 4 + ct) * 512 + lane * 16);
        acc = wmma_bf16(a, bm, acc);
      }
      #pragma unroll
      for (int i = 0; i < 8; ++i) {
        float vv = acc[i] + bv_r[i];
        int K = ct * 16 + nl;
        lv[((K >> 5) * 8 + w) * 512 + swz(hl * 8 + i, K & 31)] = f2bf(vv);
      }
    }
    // ---- forget gate (Wg rows 0..C): per-channel running sums ----
    #pragma unroll
    for (int ct = 0; ct < 4; ++ct) {
      v8f acc = zero;
      #pragma unroll
      for (int kk = 0; kk < 4; ++kk) {
        v16bf a  = *(const v16bf*)(wgA + (w * 4 + kk) * 512 + lane * 16);
        v16bf bm = *(const v16bf*)(lx + (kk * 4 + ct) * 512 + lane * 16);
        acc = wmma_bf16(a, bm, acc);
      }
      #pragma unroll
      for (int i = 0; i < 8; ++i) fs[i] += sigmoidf_(acc[i] + bgF_r[i]);
    }
    __syncthreads();

    // ---- knowledge partial: kacc[dt] += kg(16x64) x v^T(64x16) ----
    #pragma unroll
    for (int dt = 0; dt < 8; ++dt) {
      #pragma unroll
      for (int kt = 0; kt < 2; ++kt) {
        v16bf a  = *(const v16bf*)(lkg + (w * 2 + kt) * 512 + lane * 16);
        v16bf bm = *(const v16bf*)(lv + (kt * 8 + dt) * 512 + lane * 16);
        kacc[dt] = wmma_bf16(a, bm, kacc[dt]);
      }
    }
    __syncthreads();
  }

  // forget sums: butterfly over 16 columns, then one atomic per channel
  #pragma unroll
  for (int i = 0; i < 8; ++i) {
    float s = fs[i];
    s += __shfl_xor(s, 1);
    s += __shfl_xor(s, 2);
    s += __shfl_xor(s, 4);
    s += __shfl_xor(s, 8);
    if (nl == 0) atomicAdd(&fsum[b * C_DIM + w * 16 + hl * 8 + i], s);
  }
  // knowledge partials -> L2-resident global accumulator
  #pragma unroll
  for (int dt = 0; dt < 8; ++dt) {
    #pragma unroll
    for (int i = 0; i < 8; ++i) {
      atomicAdd(&know[((size_t)b * C_DIM + w * 16 + hl * 8 + i) * C_DIM + dt * 16 + nl],
                kacc[dt][i]);
    }
  }
}

// ---------------------------------------------------------------------------
// k2: new_memory = mean(forget)*prev + knowledge; write f32 result to d_out
// tail and a bf16 transposed/A-swizzled copy for the retrieval GEMM.
// ---------------------------------------------------------------------------
__global__ void __launch_bounds__(256) glak2_memory(
    const float* __restrict__ prev, const float* __restrict__ know,
    const float* __restrict__ fsum, float* __restrict__ outMem,
    unsigned short* __restrict__ memA)
{
  int idx = blockIdx.x * 256 + threadIdx.x;   // B*C*C = 262144
  int b = idx >> 14, r = idx & 16383, c = r >> 7, d = r & 127;
  float nm = fsum[b * C_DIM + c] * (1.0f / 16384.0f) * prev[idx] + know[idx];
  outMem[idx] = nm;
  // A-operand for retrieved = mem^T x q : A[m=d][k=c] = mem[c][d]
  memA[b * 16384 + ((d >> 4) * 4 + (c >> 5)) * 512 + swz(d & 15, c & 31)] = f2bf(nm);
}

// ---------------------------------------------------------------------------
// k3: q = Wq*x (+bq); retrieved = mem^T * q; out = Wp*retrieved + bp + x.
// One block = one batch x 128-column tile. LDS: ldsA holds x then retrieved,
// ldsB holds q, ldsM holds mem^T brought in by the Tensor Data Mover (TDM)
// overlapped with staging + the q GEMM. 96 KB static LDS.
// ---------------------------------------------------------------------------
__global__ void __launch_bounds__(256) glak3_out(
    const float* __restrict__ x, const unsigned short* __restrict__ ws16,
    const float* __restrict__ bq, const float* __restrict__ bp,
    float* __restrict__ out)
{
  __shared__ alignas(64) unsigned short ldsA[16384];  // x -> retrieved
  __shared__ alignas(64) unsigned short ldsB[16384];  // q
  __shared__ alignas(64) unsigned short ldsM[16384];  // mem^T (A-swizzled)
  const int tid  = threadIdx.x;
  const int lane = tid & 31;
  const int w    = tid >> 5;
  const int hl   = lane >> 4;
  const int nl   = lane & 15;
  const int b  = blockIdx.x >> 7;     // 128 col-tiles per batch
  const int n0 = (blockIdx.x & 127) * 128;

  const unsigned short* wqA = ws16 + 65536;
  const unsigned short* wpA = ws16 + 81920;
  const unsigned short* mA  = ws16 + (MEMA_BYTE / 2) + b * 16384;

#if HAVE_TDM
  // Kick off async DMA of this batch's 32KB mem^T block into LDS (wave 0).
  // D# group0: count=1, lds_addr, 57b global_addr, type=2 ("image").
  // D# group1: data_size=2B, tensor_dim0=tile_dim0=16384, dim1=tile_dim1=1,
  //            tensor_dim0_stride=16384, no multicast/pad/iterate.
  // Groups 2/3 (and the trailing group) unused for a <=2D tensor: zeros.
  if (w == 0) {
    unsigned long long ga = (unsigned long long)(uintptr_t)mA;
    unsigned int la = (unsigned int)(uintptr_t)(&ldsM[0]);
    v4u g0 = { 1u, la, (unsigned int)ga,
               (unsigned int)((ga >> 32) & 0x01FFFFFFu) | 0x80000000u };
    v8i g1 = { 0x00010000,                 // data_size = 1 (2 bytes)
               0x40000000,                 // tensor_dim0[15:0]=16384 in [31:16]
               0x00010000,                 // tensor_dim1 = 1
               0x40000000,                 // tile_dim0 = 16384 in [31:16]
               1,                          // tile_dim1 = 1
               16384,                      // tensor_dim0_stride[31:0]
               0, 0 };
    v4i gz4 = { 0, 0, 0, 0 };
    v8i gz8 = { 0, 0, 0, 0, 0, 0, 0, 0 };
    __builtin_amdgcn_tensor_load_to_lds(g0, g1, gz4, gz4, gz8, 0);
  }
#else
  for (int idx = tid; idx < 16384; idx += 256) ldsM[idx] = mA[idx];
#endif

  // stage x tile (128x128) as bf16 B-operand order
  for (int idx = tid; idx < 16384; idx += 256) {
    int c = idx >> 7, n = idx & 127;
    float xv = x[(size_t)(b * C_DIM + c) * NPIX + n0 + n];
    ldsA[((c >> 5) * 8 + (n >> 4)) * 512 + swz(n & 15, c & 31)] = f2bf(xv);
  }
  __syncthreads();

  v8f zero = {};
  float bq_r[8], bp_r[8];
  #pragma unroll
  for (int i = 0; i < 8; ++i) {
    int o = w * 16 + hl * 8 + i;
    bq_r[i] = bq[o];
    bp_r[i] = bp[o];
  }

  // ---- q = Wq*x + bq -> ldsB in B-operand order (K=out channel) ----
  #pragma unroll
  for (int nt = 0; nt < 8; ++nt) {
    v8f acc = zero;
    #pragma unroll
    for (int kk = 0; kk < 4; ++kk) {
      v16bf a  = *(const v16bf*)(wqA + (w * 4 + kk) * 512 + lane * 16);
      v16bf bm = *(const v16bf*)(ldsA + (kk * 8 + nt) * 512 + lane * 16);
      acc = wmma_bf16(a, bm, acc);
    }
    #pragma unroll
    for (int i = 0; i < 8; ++i) {
      int K = w * 16 + hl * 8 + i;
      ldsB[((K >> 5) * 8 + nt) * 512 + swz(nl, K & 31)] = f2bf(acc[i] + bq_r[i]);
    }
  }
#if HAVE_TDM
  if (w == 0) __builtin_amdgcn_s_wait_tensorcnt((short)0);
#endif
  __syncthreads();   // q ready; TDM completion (wave0) published to the block

  // ---- retrieved = mem^T * q -> ldsA (B-order: K=d) ----
  #pragma unroll
  for (int nt = 0; nt < 8; ++nt) {
    v8f acc = zero;
    #pragma unroll
    for (int kk = 0; kk < 4; ++kk) {
      v16bf a  = *(const v16bf*)(ldsM + (w * 4 + kk) * 512 + lane * 16);
      v16bf bm = *(const v16bf*)(ldsB + (kk * 8 + nt) * 512 + lane * 16);
      acc = wmma_bf16(a, bm, acc);
    }
    #pragma unroll
    for (int i = 0; i < 8; ++i) {
      int K = w * 16 + hl * 8 + i;
      ldsA[((K >> 5) * 8 + nt) * 512 + swz(nl, K & 31)] = f2bf(acc[i]);
    }
  }
  __syncthreads();

  // ---- out = Wp*retrieved + bp + x ----
  #pragma unroll
  for (int nt = 0; nt < 8; ++nt) {
    v8f acc = zero;
    #pragma unroll
    for (int kk = 0; kk < 4; ++kk) {
      v16bf a  = *(const v16bf*)(wpA + (w * 4 + kk) * 512 + lane * 16);
      v16bf bm = *(const v16bf*)(ldsA + (kk * 8 + nt) * 512 + lane * 16);
      acc = wmma_bf16(a, bm, acc);
    }
    #pragma unroll
    for (int i = 0; i < 8; ++i) {
      int o = w * 16 + hl * 8 + i;
      size_t gi = (size_t)(b * C_DIM + o) * NPIX + n0 + nt * 16 + nl;
      out[gi] = acc[i] + bp_r[i] + x[gi];
    }
  }
}

// ---------------------------------------------------------------------------
extern "C" void kernel_launch(void* const* d_in, const int* in_sizes, int n_in,
                              void* d_out, int out_size, void* d_ws, size_t ws_size,
                              hipStream_t stream) {
  const float* x    = (const float*)d_in[0];
  const float* prev = (const float*)d_in[1];
  const float* Wq   = (const float*)d_in[2];
  const float* bq   = (const float*)d_in[3];
  const float* Wk   = (const float*)d_in[4];
  const float* bk   = (const float*)d_in[5];
  const float* Wv   = (const float*)d_in[6];
  const float* bv   = (const float*)d_in[7];
  const float* Wg   = (const float*)d_in[8];
  const float* bg   = (const float*)d_in[9];
  const float* Wp   = (const float*)d_in[10];
  const float* bp   = (const float*)d_in[11];

  float* out = (float*)d_out;
  unsigned short* ws16 = (unsigned short*)d_ws;
  float* know = (float*)((char*)d_ws + KNOW_BYTE);
  float* fsum = (float*)((char*)d_ws + FSUM_BYTE);
  unsigned short* memA = (unsigned short*)((char*)d_ws + MEMA_BYTE);

  // zero the accumulators (know + fsum are contiguous) -- graph-capturable
  (void)hipMemsetAsync((char*)d_ws + KNOW_BYTE, 0,
                       (FSUM_BYTE - KNOW_BYTE) + BATCH * C_DIM * 4, stream);

  glak0_prep<<<384, 256, 0, stream>>>(Wq, Wk, Wv, Wg, Wp, ws16);
  glak1_proj_knowledge<<<BATCH * 64, 256, 0, stream>>>(x, ws16, bk, bv, bg, know, fsum);
  glak2_memory<<<(BATCH * C_DIM * C_DIM) / 256, 256, 0, stream>>>(
      prev, know, fsum, out + (size_t)BATCH * C_DIM * NPIX, memA);
  glak3_out<<<BATCH * 128, 256, 0, stream>>>(x, ws16, bq, bp, out);
}
